// PLE_8263517077686
// MI455X (gfx1250) — compile-verified, matrix operand order checked by reading, not run on hardware
//
#include <hip/hip_runtime.h>
#include <math.h>

// ---------------------------------------------------------------------------
// PLE multi-task model, fused bf16-WMMA pipeline for gfx1250 (MI455X).
// GEMMs run on v_wmma_f32_16x16x32_bf16 (f32 accumulate). The shared B tile
// (64 cols x 32 K) is staged into LDS by the Tensor Data Mover
// (tensor_load_to_lds, TENSORcnt-tracked, double buffered, TDM row padding
// to avoid LDS bank conflicts); A streams from global with WGP-scope
// prefetch. Falls back to direct global B loads if the TDM builtin is absent.
// ---------------------------------------------------------------------------

#define B_    16384
#define F_    26
#define V_    100000
#define ED_   32
#define IND_  832
#define EXD_  256
#define NSE_  4
#define NTE_  2
#define NG_   6
#define EPS_  1e-5f

typedef __bf16 bf16;
typedef __bf16  bf16x8  __attribute__((ext_vector_type(8)));
typedef __bf16  bf16x16 __attribute__((ext_vector_type(16)));
typedef float   f32x8   __attribute__((ext_vector_type(8)));
typedef unsigned short u16;   // bf16 storage type across host/device interface

#if __has_builtin(__builtin_amdgcn_tensor_load_to_lds) && \
    __has_builtin(__builtin_amdgcn_s_wait_tensorcnt)
#define USE_TDM 1
typedef unsigned int uint32x4 __attribute__((ext_vector_type(4)));
typedef int          int32x8  __attribute__((ext_vector_type(8)));
typedef int          int32x4  __attribute__((ext_vector_type(4)));
#else
#define USE_TDM 0
#endif

// ---------------------------------------------------------------------------
// Embedding gather: e[b, f*32+d] = (bf16) emb[f, x[b,f], d]
// ---------------------------------------------------------------------------
__global__ void k_embed_gather(const int* __restrict__ x,
                               const float* __restrict__ emb,
                               u16* __restrict__ e) {
  long i = (long)blockIdx.x * blockDim.x + threadIdx.x;
  if (i >= (long)B_ * F_ * ED_) return;
  int  d = (int)(i & (ED_ - 1));
  long t = i >> 5;
  int  f = (int)(t % F_);
  long b = t / F_;
  int  row = x[b * F_ + f];
  float val = emb[((long)f * V_ + row) * ED_ + d];
  reinterpret_cast<bf16*>(e)[b * IND_ + f * ED_ + d] = (bf16)val;
}

// ---------------------------------------------------------------------------
// Weight prep: w [KE, D, E] f32 row-major  ->  wT [KE, E, D] bf16 (N-major)
// ---------------------------------------------------------------------------
__global__ void k_prep_wT(const float* __restrict__ w, u16* __restrict__ wT,
                          int KE, int D, int E) {
  long i = (long)blockIdx.x * blockDim.x + threadIdx.x;
  long total = (long)KE * D * E;
  if (i >= total) return;
  int  e = (int)(i % E);
  long t = i / E;
  int  d = (int)(t % D);
  int  k = (int)(t / D);
  reinterpret_cast<bf16*>(wT)[((long)k * E + e) * D + d] = (bf16)w[i];
}

// ---------------------------------------------------------------------------
// TDM descriptor issue: copy a [64 rows x 32 elem] bf16 tile (rows are rows
// of BT, i.e. output columns; contiguous K in dim0) into LDS, padding each
// 64B row with 16B (pad_interval=16 DWORDs, pad_amount=4 DWORDs) so LDS row
// stride is 80B = 40 u16 = 20 banks.
// D# packing per CDNA5 ISA ch.8: group0 {count=1, lds_addr, global_addr,
// type=2}; group1 {data_size=1(2B), pad_en, tensor_dim0=K, tensor_dim1=N,
// tile_dim0=32, tile_dim1=64, tensor_dim0_stride=K}. Groups 2/3 and the
// trailing group are zero (2-D tensor, no gather/iterate).
// ---------------------------------------------------------------------------
#if USE_TDM
__device__ __forceinline__ void tdm_load_b_tile(const bf16* src,
                                                unsigned lds_off,
                                                int K, int N) {
  unsigned long long ga = (unsigned long long)(const void*)src;
  uint32x4 g0;
  g0[0] = 1u;                                   // count=1, user descriptor
  g0[1] = lds_off;                              // lds_addr [63:32]
  g0[2] = (unsigned)(ga & 0xFFFFFFFFu);         // global_addr [95:64]
  g0[3] = (unsigned)((ga >> 32) & 0x01FFFFFFu)  // global_addr [120:96]
          | (2u << 30);                         // type=2 ("image") [127:126]
  int32x8 g1;
  g1[0] = (1 << 16)        // data_size = 1 -> 2 bytes
        | (1 << 20)        // pad_enable
        | (3 << 22)        // pad_interval code 3 -> every 16 DWORDs (64B)
        | (3 << 25);       // pad_amount  code 3 -> 4 DWORDs (16B)
  g1[1] = (int)((unsigned)K << 16);   // tensor_dim0[15:0]  at bits 63:48
  g1[2] = (int)((unsigned)N << 16);   // tensor_dim0[31:16]=0, tensor_dim1 lo
  g1[3] = (int)(32u << 16);           // tensor_dim1 hi = 0, tile_dim0 = 32
  g1[4] = 64;                          // tile_dim1 = 64, tile_dim2 = 0
  g1[5] = K;                           // tensor_dim0_stride low 32
  g1[6] = 0;                           // stride hi, dim1_stride lo
  g1[7] = 0;
  int32x4 z4;
  z4[0] = 0; z4[1] = 0; z4[2] = 0; z4[3] = 0;
  int32x8 z8;
  z8[0] = 0; z8[1] = 0; z8[2] = 0; z8[3] = 0;
  z8[4] = 0; z8[5] = 0; z8[6] = 0; z8[7] = 0;
  __builtin_amdgcn_tensor_load_to_lds(g0, g1, z4, z4, z8, 0);
}
#endif

// ---------------------------------------------------------------------------
// WMMA GEMM with fused bias + BatchNorm(inference) + ReLU, bf16 in/out,
// f32 accumulate.  C[z] = act( (A[z] @ B[z]^T + bias) bn ) as [M, N] bf16.
// Block = 8 waves (256 thr) -> 128 rows x 64 cols per block; each wave owns
// a 16x64 tile (4 accumulators, A fragment reused 4x). B tile staged in LDS
// by TDM (double buffered); A fragments stream from global.
// 16-bit A-matrix 16x32 lane layout per CDNA5 ISA §7.12.2:
//   lanes 0-15 : row = lane,    K = {0..7} then {16..23}
//   lanes 16-31: row = lane-16, K = {8..15} then {24..31}
// ---------------------------------------------------------------------------
#define GEMM_WAVES 8
#define LDS_ROW    40   // 32 elems + 4-DWORD TDM pad = 80B stride

__global__ void k_wmma_gemm(const u16* __restrict__ Au, long aZ,
                            const u16* __restrict__ Bu, long bZ,
                            const float* __restrict__ bias,
                            const float* __restrict__ bn_g,
                            const float* __restrict__ bn_b,
                            const float* __restrict__ bn_m,
                            const float* __restrict__ bn_v,
                            long pZ,
                            u16* __restrict__ Cu, long cZ,
                            int M, int N, int K, int do_relu) {
  const bf16* A  = reinterpret_cast<const bf16*>(Au) + (long)blockIdx.z * aZ;
  const bf16* BT = reinterpret_cast<const bf16*>(Bu) + (long)blockIdx.z * bZ;
  bf16*       C  = reinterpret_cast<bf16*>(Cu)       + (long)blockIdx.z * cZ;
  long po = (long)blockIdx.z * pZ;

  int wave = threadIdx.x >> 5;
  int lane = threadIdx.x & 31;
  int row0 = (blockIdx.x * GEMM_WAVES + wave) * 16;
  int col0 = blockIdx.y * 64;
  int fr   = lane & 15;          // row within fragment (M for A, N for B)
  int kh   = (lane >> 4) << 3;   // K phase: 0 or 8

  const bf16* ap = A + (long)(row0 + fr) * K + kh;

  union Frag { bf16x16 v; bf16x8 h[2]; };
  f32x8 acc[4] = {};

#if USE_TDM
  __shared__ __align__(16) u16 sB[2][64][LDS_ROW];
  const bool leader = (threadIdx.x < 32);
  const bf16* btile = BT + (long)col0 * K;
  unsigned lds_base[2] = {
    (unsigned)(unsigned long long)(const void*)&sB[0][0][0],
    (unsigned)(unsigned long long)(const void*)&sB[1][0][0]
  };
  if (leader) tdm_load_b_tile(btile, lds_base[0], K, N);

  int nk = K >> 5;
  for (int s = 0; s < nk; ++s) {
    int k   = s << 5;
    int cur = s & 1;
    if (leader) {
      if (s + 1 < nk) {
        tdm_load_b_tile(btile + (k + 32), lds_base[cur ^ 1], K, N);
        __builtin_amdgcn_s_wait_tensorcnt(1);   // previous (cur) tile done
      } else {
        __builtin_amdgcn_s_wait_tensorcnt(0);
      }
    }
    __syncthreads();                            // sB[cur] visible to all waves

    Frag a;
    a.h[0] = *(const bf16x8*)(ap + k);
    a.h[1] = *(const bf16x8*)(ap + k + 16);
    __builtin_prefetch(ap + k + 512, 0, 3);     // WGP-scope global_prefetch_b8
#pragma unroll
    for (int j = 0; j < 4; ++j) {
      const bf16* bp = (const bf16*)&sB[cur][j * 16 + fr][0];   // ds_load_b128
      Frag b;
      b.h[0] = *(const bf16x8*)(bp + kh);
      b.h[1] = *(const bf16x8*)(bp + 16 + kh);
      acc[j] = __builtin_amdgcn_wmma_f32_16x16x32_bf16(
          false, a.v, false, b.v, (short)0, acc[j], false, false);
    }
    __syncthreads();                            // done reading sB[cur]
  }
#else
  const bf16* bp0 = BT + (long)(col0 + fr) * K + kh;
  for (int k = 0; k < K; k += 32) {
    Frag a;
    a.h[0] = *(const bf16x8*)(ap + k);
    a.h[1] = *(const bf16x8*)(ap + k + 16);
    __builtin_prefetch(ap + k + 512, 0, 3);
#pragma unroll
    for (int j = 0; j < 4; ++j) {
      const bf16* bp = bp0 + (long)j * 16 * K;
      Frag b;
      b.h[0] = *(const bf16x8*)(bp + k);
      b.h[1] = *(const bf16x8*)(bp + k + 16);
      acc[j] = __builtin_amdgcn_wmma_f32_16x16x32_bf16(
          false, a.v, false, b.v, (short)0, acc[j], false, false);
    }
  }
#endif

  // C/D f32 layout: lanes 0-15 -> M = row0 + r, lanes 16-31 -> M = row0+8+r,
  // N = col0 + 16*j + (lane & 15).
  int orow = row0 + ((lane >> 4) << 3);
#pragma unroll
  for (int j = 0; j < 4; ++j) {
    int col = col0 + j * 16 + fr;
    float bv = bias ? bias[po + col] : 0.0f;
    float scale = 1.0f, shift = 0.0f;
    if (bn_g) {
      float inv = bn_g[po + col] * rsqrtf(bn_v[po + col] + EPS_);
      scale = inv;
      shift = bn_b[po + col] - bn_m[po + col] * inv;
    }
#pragma unroll
    for (int r = 0; r < 8; ++r) {
      float val = (acc[j][r] + bv) * scale + shift;
      if (do_relu) val = fmaxf(val, 0.0f);
      C[(long)(orow + r) * N + col] = (bf16)val;
    }
  }
}

// ---------------------------------------------------------------------------
// Gate: softmax(x @ gw + gb) over NG_=6 experts, per row. One thread per row.
// ---------------------------------------------------------------------------
__global__ void k_gate(const u16* __restrict__ xu,
                       const float* __restrict__ gw,
                       const float* __restrict__ gb,
                       float* __restrict__ gout, int D) {
  int b = blockIdx.x * blockDim.x + threadIdx.x;
  if (b >= B_) return;
  const bf16* x = reinterpret_cast<const bf16*>(xu) + (long)b * D;
  float s[NG_];
#pragma unroll
  for (int k = 0; k < NG_; ++k) s[k] = gb[k];
  for (int d = 0; d < D; ++d) {
    float xv = (float)x[d];
#pragma unroll
    for (int k = 0; k < NG_; ++k) s[k] += xv * gw[(long)d * NG_ + k];
  }
  float mx = s[0];
#pragma unroll
  for (int k = 1; k < NG_; ++k) mx = fmaxf(mx, s[k]);
  float den = 0.0f;
#pragma unroll
  for (int k = 0; k < NG_; ++k) { s[k] = expf(s[k] - mx); den += s[k]; }
  float inv = 1.0f / den;
#pragma unroll
  for (int k = 0; k < NG_; ++k) gout[(long)b * NG_ + k] = s[k] * inv;
}

// ---------------------------------------------------------------------------
// Gated mix: out[b,e] = sum_k g[b,k] * concat(shared[4], task[2])[k,b,e]
// ---------------------------------------------------------------------------
__global__ void k_mix(const u16* __restrict__ sh, const u16* __restrict__ tk,
                      const float* __restrict__ gate, u16* __restrict__ out) {
  long i = (long)blockIdx.x * blockDim.x + threadIdx.x;
  if (i >= (long)B_ * EXD_) return;
  long b = i >> 8;
  int  e = (int)(i & (EXD_ - 1));
  const float* g = gate + b * NG_;
  const bf16* S = reinterpret_cast<const bf16*>(sh);
  const bf16* T = reinterpret_cast<const bf16*>(tk);
  float s = 0.0f;
#pragma unroll
  for (int k = 0; k < NSE_; ++k) s += g[k] * (float)S[((long)k * B_ + b) * EXD_ + e];
#pragma unroll
  for (int k = 0; k < NTE_; ++k) s += g[NSE_ + k] * (float)T[((long)k * B_ + b) * EXD_ + e];
  reinterpret_cast<bf16*>(out)[i] = (bf16)s;
}

// ---------------------------------------------------------------------------
// Final: per-row 128-dot + bias, sigmoid, write cp, vp, cp*vp.
// ---------------------------------------------------------------------------
__global__ void k_final(const u16* __restrict__ h0u, const u16* __restrict__ h1u,
                        const float* __restrict__ wo0, const float* __restrict__ bo0,
                        const float* __restrict__ wo1, const float* __restrict__ bo1,
                        float* __restrict__ out) {
  int b = blockIdx.x * blockDim.x + threadIdx.x;
  if (b >= B_) return;
  const bf16* h0 = reinterpret_cast<const bf16*>(h0u) + (long)b * 128;
  const bf16* h1 = reinterpret_cast<const bf16*>(h1u) + (long)b * 128;
  float s0 = bo0[0], s1 = bo1[0];
#pragma unroll 4
  for (int i = 0; i < 128; ++i) {
    s0 += (float)h0[i] * wo0[i];
    s1 += (float)h1[i] * wo1[i];
  }
  float cp = 1.0f / (1.0f + expf(-s0));
  float vp = 1.0f / (1.0f + expf(-s1));
  out[b] = cp;
  out[B_ + b] = vp;
  out[2 * B_ + b] = cp * vp;
}

// ---------------------------------------------------------------------------
// Host orchestration
// ---------------------------------------------------------------------------
struct ExpertP { const float *w1, *b1, *g, *bt, *m, *v, *w2, *b2; };
struct LayerP  { ExpertP sh, t0, t1; const float *gw, *gb; };
struct TowerP  { const float *w0, *w1, *b0, *b1, *g0, *g1, *bt0, *bt1,
                             *m0, *m1, *v0, *v1, *wo, *bo; };

static ExpertP grab_expert(void* const* d_in, int& i) {
  ExpertP e;
  e.w1 = (const float*)d_in[i++]; e.b1 = (const float*)d_in[i++];
  e.g  = (const float*)d_in[i++]; e.bt = (const float*)d_in[i++];
  e.m  = (const float*)d_in[i++]; e.v  = (const float*)d_in[i++];
  e.w2 = (const float*)d_in[i++]; e.b2 = (const float*)d_in[i++];
  return e;
}

extern "C" void kernel_launch(void* const* d_in, const int* in_sizes, int n_in,
                              void* d_out, int out_size, void* d_ws, size_t ws_size,
                              hipStream_t stream) {
  (void)in_sizes; (void)n_in; (void)out_size; (void)ws_size;

  int i = 0;
  const int*   x   = (const int*)d_in[i++];
  const float* emb = (const float*)d_in[i++];
  LayerP L[2];
  for (int l = 0; l < 2; ++l) {
    L[l].sh = grab_expert(d_in, i);
    L[l].t0 = grab_expert(d_in, i);
    L[l].t1 = grab_expert(d_in, i);
    L[l].gw = (const float*)d_in[i++];
    L[l].gb = (const float*)d_in[i++];
  }
  TowerP T[2];
  for (int t = 0; t < 2; ++t) {
    T[t].w0  = (const float*)d_in[i++]; T[t].w1  = (const float*)d_in[i++];
    T[t].b0  = (const float*)d_in[i++]; T[t].b1  = (const float*)d_in[i++];
    T[t].g0  = (const float*)d_in[i++]; T[t].g1  = (const float*)d_in[i++];
    T[t].bt0 = (const float*)d_in[i++]; T[t].bt1 = (const float*)d_in[i++];
    T[t].m0  = (const float*)d_in[i++]; T[t].m1  = (const float*)d_in[i++];
    T[t].v0  = (const float*)d_in[i++]; T[t].v1  = (const float*)d_in[i++];
    T[t].wo  = (const float*)d_in[i++]; T[t].bo  = (const float*)d_in[i++];
  }

  // Workspace bump allocator (256B aligned).
  char* wsp = (char*)d_ws;
  size_t off = 0;
  auto alloc = [&](size_t bytes) -> void* {
    void* p = wsp + off;
    off = (off + bytes + 255) & ~(size_t)255;
    return p;
  };

  const int Dl[2] = { IND_, EXD_ };

  u16* e_bf = (u16*)alloc((size_t)B_ * IND_ * 2);

  u16 *w1T[2][3], *w2T[2][3];   // [layer][shared, task0, task1]
  for (int l = 0; l < 2; ++l) {
    w1T[l][0] = (u16*)alloc((size_t)NSE_ * EXD_ * Dl[l] * 2);
    w1T[l][1] = (u16*)alloc((size_t)NTE_ * EXD_ * Dl[l] * 2);
    w1T[l][2] = (u16*)alloc((size_t)NTE_ * EXD_ * Dl[l] * 2);
    w2T[l][0] = (u16*)alloc((size_t)NSE_ * EXD_ * EXD_ * 2);
    w2T[l][1] = (u16*)alloc((size_t)NTE_ * EXD_ * EXD_ * 2);
    w2T[l][2] = (u16*)alloc((size_t)NTE_ * EXD_ * EXD_ * 2);
  }
  u16 *tw0T[2], *tw1T[2];
  for (int t = 0; t < 2; ++t) {
    tw0T[t] = (u16*)alloc((size_t)EXD_ * EXD_ * 2);
    tw1T[t] = (u16*)alloc((size_t)128 * EXD_ * 2);
  }

  u16* h_sh   = (u16*)alloc((size_t)NSE_ * B_ * EXD_ * 2);
  u16* h_tk   = (u16*)alloc((size_t)NTE_ * B_ * EXD_ * 2);
  u16* out_sh = (u16*)alloc((size_t)NSE_ * B_ * EXD_ * 2);
  u16* out_tk = (u16*)alloc((size_t)NTE_ * B_ * EXD_ * 2);
  u16* tiA[2] = { (u16*)alloc((size_t)B_ * EXD_ * 2), (u16*)alloc((size_t)B_ * EXD_ * 2) };
  u16* tiB[2] = { (u16*)alloc((size_t)B_ * EXD_ * 2), (u16*)alloc((size_t)B_ * EXD_ * 2) };
  float* gbuf = (float*)alloc((size_t)B_ * NG_ * 4);
  u16* th1    = (u16*)alloc((size_t)B_ * EXD_ * 2);
  u16* th2[2] = { (u16*)alloc((size_t)B_ * 128 * 2), (u16*)alloc((size_t)B_ * 128 * 2) };

  auto prep = [&](const float* w, u16* dst, int KE, int D, int E) {
    long total = (long)KE * D * E;
    k_prep_wT<<<dim3((unsigned)((total + 255) / 256)), dim3(256), 0, stream>>>(
        w, dst, KE, D, E);
  };
  auto gemm = [&](const u16* A, long aZ, const u16* BT, long bZ,
                  const float* bias, const float* g, const float* bt,
                  const float* m, const float* v, long pZ,
                  u16* C, long cZ, int M, int N, int K, int relu, int Z) {
    dim3 grid((unsigned)(M / (16 * GEMM_WAVES)), (unsigned)(N / 64), (unsigned)Z);
    k_wmma_gemm<<<grid, dim3(32 * GEMM_WAVES), 0, stream>>>(
        A, aZ, BT, bZ, bias, g, bt, m, v, pZ, C, cZ, M, N, K, relu);
  };

  // 1) Embedding gather -> bf16 e
  {
    long total = (long)B_ * F_ * ED_;
    k_embed_gather<<<dim3((unsigned)((total + 255) / 256)), dim3(256), 0, stream>>>(
        x, emb, e_bf);
  }

  // 2) Weight prep (f32 -> bf16, transposed to [N, K])
  for (int l = 0; l < 2; ++l) {
    prep(L[l].sh.w1, w1T[l][0], NSE_, Dl[l], EXD_);
    prep(L[l].t0.w1, w1T[l][1], NTE_, Dl[l], EXD_);
    prep(L[l].t1.w1, w1T[l][2], NTE_, Dl[l], EXD_);
    prep(L[l].sh.w2, w2T[l][0], NSE_, EXD_, EXD_);
    prep(L[l].t0.w2, w2T[l][1], NTE_, EXD_, EXD_);
    prep(L[l].t1.w2, w2T[l][2], NTE_, EXD_, EXD_);
  }
  for (int t = 0; t < 2; ++t) {
    prep(T[t].w0, tw0T[t], 1, EXD_, EXD_);
    prep(T[t].w1, tw1T[t], 1, EXD_, 128);
  }

  // 3) PLE layers
  const u16* in0 = e_bf;
  const u16* in1 = e_bf;
  for (int l = 0; l < 2; ++l) {
    int D = Dl[l];
    long hz = (long)B_ * EXD_;
    u16** tout = (l == 0) ? tiA : tiB;
    const ExpertP* tsk[2] = { &L[l].t0, &L[l].t1 };
    const u16* tin[2] = { in0, in1 };
    u16* tw1l[2] = { w1T[l][1], w1T[l][2] };
    u16* tw2l[2] = { w2T[l][1], w2T[l][2] };

    // shared experts (input = task-0 input, as in reference)
    gemm(in0, 0, w1T[l][0], (long)EXD_ * D,
         L[l].sh.b1, L[l].sh.g, L[l].sh.bt, L[l].sh.m, L[l].sh.v, EXD_,
         h_sh, hz, B_, EXD_, D, 1, NSE_);
    gemm(h_sh, hz, w2T[l][0], (long)EXD_ * EXD_,
         L[l].sh.b2, nullptr, nullptr, nullptr, nullptr, EXD_,
         out_sh, hz, B_, EXD_, EXD_, 1, NSE_);

    for (int t = 0; t < 2; ++t) {
      const ExpertP* E = tsk[t];
      gemm(tin[t], 0, tw1l[t], (long)EXD_ * D,
           E->b1, E->g, E->bt, E->m, E->v, EXD_,
           h_tk, hz, B_, EXD_, D, 1, NTE_);
      gemm(h_tk, hz, tw2l[t], (long)EXD_ * EXD_,
           E->b2, nullptr, nullptr, nullptr, nullptr, EXD_,
           out_tk, hz, B_, EXD_, EXD_, 1, NTE_);
      k_gate<<<dim3(B_ / 256), dim3(256), 0, stream>>>(
          tin[t], L[l].gw + (long)t * D * NG_, L[l].gb + (long)t * NG_, gbuf, D);
      k_mix<<<dim3((B_ * EXD_) / 256), dim3(256), 0, stream>>>(
          out_sh, out_tk, gbuf, tout[t]);
    }
    in0 = tout[0];
    in1 = tout[1];
  }

  // 4) Towers
  const u16* tower_in[2] = { in0, in1 };
  for (int t = 0; t < 2; ++t) {
    gemm(tower_in[t], 0, tw0T[t], 0,
         T[t].b0, T[t].g0, T[t].bt0, T[t].m0, T[t].v0, 0,
         th1, 0, B_, EXD_, EXD_, 1, 1);
    gemm(th1, 0, tw1T[t], 0,
         T[t].b1, T[t].g1, T[t].bt1, T[t].m1, T[t].v1, 0,
         th2[t], 0, B_, 128, EXD_, 1, 1);
  }

  // 5) Heads: sigmoid + product
  k_final<<<dim3(B_ / 256), dim3(256), 0, stream>>>(
      th2[0], th2[1], T[0].wo, T[0].bo, T[1].wo, T[1].bo, (float*)d_out);
}
